// RAN_O_5222680232619
// MI455X (gfx1250) — matrix operand order repaired and checked by reading, not compiled, not run on hardware
//
#include <hip/hip_runtime.h>
#include <hip/hip_bf16.h>

typedef __attribute__((ext_vector_type(16))) _Float16 v16h;
typedef __attribute__((ext_vector_type(8)))  float    v8f;
typedef __attribute__((ext_vector_type(4)))  int      vi4;
typedef __attribute__((ext_vector_type(4)))  float    vf4;

#define NB 8
#define NN 2048
#define ND 64
#define NEG_SLOPE 0.2f

// ---------------------------------------------------------------------------
// Phase 1: h = feats @ W^T  (per row), a_src = h.attn_src, a_dst = h.attn_dst.
// h is stored in f16 arranged as WMMA B-matrix (32x16, f16) fragments:
//   frag = (b*64 + j/32)*4 + d/16
//   lane = (d%16) + ((j%32) & 8 ? 16 : 0),  elem = (j%8) + (j%32 >= 16 ? 8 : 0)   (ISA 7.12.2 B layout)
//   hbuf[frag*512 + lane*16 + elem]
// ---------------------------------------------------------------------------
__global__ void __launch_bounds__(256)
gat_phase1(const float* __restrict__ feats, const float* __restrict__ W,
           const float* __restrict__ aw_src, const float* __restrict__ aw_dst,
           _Float16* __restrict__ hbuf, float* __restrict__ a_src,
           float* __restrict__ a_dst)
{
    __shared__ float Wt[64 * 65];   // transposed + padded: Wt[d*65+o] = W[o*64+d]
    __shared__ float fr[4 * 64];
    __shared__ float ssrc[4], sdst[4];

    const int tid = threadIdx.x;
    const int rowbase = blockIdx.x * 4;

    for (int k = tid; k < 64 * 64; k += 256) {
        int o = k >> 6, d = k & 63;
        Wt[d * 65 + o] = W[k];
    }
    {
        int r = tid >> 6, d = tid & 63;
        fr[tid] = feats[(size_t)(rowbase + r) * ND + d];
    }
    if (tid < 4) { ssrc[tid] = 0.f; sdst[tid] = 0.f; }
    __syncthreads();

    const int r = tid >> 6, o = tid & 63;
    float acc = 0.f;
#pragma unroll
    for (int d = 0; d < 64; ++d)
        acc = fmaf(fr[r * 64 + d], Wt[d * 65 + o], acc);

    atomicAdd(&ssrc[r], acc * aw_src[o]);   // ds_add_f32
    atomicAdd(&sdst[r], acc * aw_dst[o]);

    const int row = rowbase + r;
    const int bb = row >> 11;          // /2048
    const int j  = row & 2047;
    const int frag = (bb * 64 + (j >> 5)) * 4 + (o >> 4);
    const int jj = j & 31;
    const int lane = (o & 15) + ((jj & 8) ? 16 : 0);
    hbuf[(size_t)frag * 512 + lane * 16 + ((jj & 7) | ((jj >> 1) & 8))] = (_Float16)acc;

    __syncthreads();
    if (tid < 4) {
        a_src[rowbase + tid] = ssrc[tid];
        a_dst[rowbase + tid] = sdst[tid];
    }
}

// ---------------------------------------------------------------------------
// Phase 2: streaming masked softmax-aggregation, flash-style, no rescale:
// m_i = leaky(a_src[i] + max_j a_dst[j]) bounds every logit in the row.
// Block = 16 rows x all 2048 cols; 4 waves split the 64 j-tiles (K=32 each).
// Mask loads are non-temporal (268 MB stream > 192 MB L2) so the reused
// h fragments / a_dst stay L2-resident.
// ---------------------------------------------------------------------------
__global__ void __launch_bounds__(128, 4)
gat_phase2(const int* __restrict__ madjg, const int* __restrict__ mjobg,
           const long long* __restrict__ bidx,
           const float* __restrict__ feats, const float* __restrict__ lam,
           const _Float16* __restrict__ hbuf,
           const float* __restrict__ a_src, const float* __restrict__ a_dst,
           float* __restrict__ out)
{
    __shared__ float accs[2][16][64];     // 8 KB cross-wave accumulator tile
    __shared__ float sums[2][16];
    __shared__ float asrc_s[16], m_s[16];
    __shared__ float red[128];

    const int tid = threadIdx.x;
    const int b = blockIdx.x >> 7;              // 128 row-tiles per batch
    const int rowbase = (blockIdx.x & 127) << 4;
    const int mb = (int)bidx[b];                // mask gather index

    for (int k = tid; k < 2 * 16 * 64; k += 128) ((float*)accs)[k] = 0.f;
    if (tid < 32) ((float*)sums)[tid] = 0.f;

    const float* ad = a_dst + (size_t)b * NN;
    float mx = -3.0e38f;
    for (int k = tid; k < NN; k += 128) mx = fmaxf(mx, ad[k]);
    red[tid] = mx;
    __syncthreads();
    for (int s = 64; s > 0; s >>= 1) {
        if (tid < s) red[tid] = fmaxf(red[tid], red[tid + s]);
        __syncthreads();
    }
    if (tid < 16) {
        float as = a_src[(size_t)b * NN + rowbase + tid];
        asrc_s[tid] = as;
        float mm = as + red[0];
        m_s[tid] = (mm >= 0.f) ? mm : NEG_SLOPE * mm;
    }
    __syncthreads();

    const int wave = tid >> 5, L = tid & 31;
    const int iloc = L & 15;
    const bool hi = (L >= 16);
    const float a_s = asrc_s[iloc];
    const float m_i = m_s[iloc];

    const int* mA = madjg + ((size_t)mb * NN + rowbase + iloc) * NN;
    const int* mJ = mjobg + ((size_t)mb * NN + rowbase + iloc) * NN;

    v8f accA[4] = {}, accJ[4] = {};
    float sA = 0.f, sJ = 0.f;

    for (int t = wave; t < 64; t += 4) {
        // A-matrix 16x32 f16 layout: lane<16 holds K{0..7,16..23},
        // lane>=16 holds K{8..15,24..31} for row M = lane%16.
        const int j0 = (t << 5) + (hi ? 8 : 0);

        vf4 q0 = *(const vf4*)(ad + j0);
        vf4 q1 = *(const vf4*)(ad + j0 + 4);
        vf4 q2 = *(const vf4*)(ad + j0 + 16);
        vf4 q3 = *(const vf4*)(ad + j0 + 20);
        vi4 x0 = __builtin_nontemporal_load((const vi4*)(mA + j0));
        vi4 x1 = __builtin_nontemporal_load((const vi4*)(mA + j0 + 4));
        vi4 x2 = __builtin_nontemporal_load((const vi4*)(mA + j0 + 16));
        vi4 x3 = __builtin_nontemporal_load((const vi4*)(mA + j0 + 20));
        vi4 y0 = __builtin_nontemporal_load((const vi4*)(mJ + j0));
        vi4 y1 = __builtin_nontemporal_load((const vi4*)(mJ + j0 + 4));
        vi4 y2 = __builtin_nontemporal_load((const vi4*)(mJ + j0 + 16));
        vi4 y3 = __builtin_nontemporal_load((const vi4*)(mJ + j0 + 20));

        float dv[16] = { q0.x, q0.y, q0.z, q0.w, q1.x, q1.y, q1.z, q1.w,
                         q2.x, q2.y, q2.z, q2.w, q3.x, q3.y, q3.z, q3.w };
        int ma[16] = { x0.x, x0.y, x0.z, x0.w, x1.x, x1.y, x1.z, x1.w,
                       x2.x, x2.y, x2.z, x2.w, x3.x, x3.y, x3.z, x3.w };
        int mj[16] = { y0.x, y0.y, y0.z, y0.w, y1.x, y1.y, y1.z, y1.w,
                       y2.x, y2.y, y2.z, y2.w, y3.x, y3.y, y3.z, y3.w };

        v16h fa, fj;
#pragma unroll
        for (int e = 0; e < 16; ++e) {
            float eij = a_s + dv[e];
            float lk  = (eij >= 0.f) ? eij : NEG_SLOPE * eij;
            float pex = __expf(lk - m_i);          // <= 1 by construction
            float pa  = (ma[e] == 1) ? pex : 0.f;
            float pj  = (mj[e] == 1) ? pex : 0.f;
            sA += pa; sJ += pj;
            fa[e] = (_Float16)pa;
            fj[e] = (_Float16)pj;
        }

        const _Float16* hb = hbuf + (((size_t)b * 64 + t) << 11) + (size_t)L * 16;
#pragma unroll
        for (int dt = 0; dt < 4; ++dt) {
            v16h bf = *(const v16h*)(hb + (dt << 9));
            accA[dt] = __builtin_amdgcn_wmma_f32_16x16x32_f16(
                false, fa, false, bf, (short)0, accA[dt], false, false);
            accJ[dt] = __builtin_amdgcn_wmma_f32_16x16x32_f16(
                false, fj, false, bf, (short)0, accJ[dt], false, false);
        }
    }

    atomicAdd(&sums[0][iloc], sA);
    atomicAdd(&sums[1][iloc], sJ);
#pragma unroll
    for (int dt = 0; dt < 4; ++dt) {
#pragma unroll
        for (int v = 0; v < 8; ++v) {
            int rw = v + (hi ? 8 : 0);             // C layout: lane>=16 -> M+8
            atomicAdd(&accs[0][rw][dt * 16 + iloc], accA[dt][v]);
            atomicAdd(&accs[1][rw][dt * 16 + iloc], accJ[dt][v]);
        }
    }
    __syncthreads();

    for (int k = tid; k < 16 * 64; k += 128) {
        int rw = k >> 6, d = k & 63;
        float l0 = __expf(lam[2 * d]);
        float l1 = __expf(lam[2 * d + 1]);
        float inv = 1.0f / (l0 + l1);
        float sa = fmaxf(sums[0][rw], 1e-30f);
        float sj = fmaxf(sums[1][rw], 1e-30f);
        size_t oi = ((size_t)b * NN + rowbase + rw) * ND + d;
        float v = (l0 * inv) * accs[0][rw][d] / sa
                + (l1 * inv) * accs[1][rw][d] / sj
                + feats[oi];
        __builtin_nontemporal_store(v, &out[oi]);  // write-once, keep out of L2
    }
}

extern "C" void kernel_launch(void* const* d_in, const int* in_sizes, int n_in,
                              void* d_out, int out_size, void* d_ws, size_t ws_size,
                              hipStream_t stream) {
    const int*       madj  = (const int*)d_in[0];
    const int*       mjob  = (const int*)d_in[1];
    const long long* bidx  = (const long long*)d_in[2];
    const float*     feats = (const float*)d_in[3];
    const float*     W     = (const float*)d_in[4];
    const float*     asw   = (const float*)d_in[5];
    const float*     adw   = (const float*)d_in[6];
    const float*     lam   = (const float*)d_in[7];
    float* out = (float*)d_out;

    _Float16* hbuf  = (_Float16*)d_ws;                              // 2 MB
    float*    asrc  = (float*)((char*)d_ws + (size_t)NB * NN * ND * 2);
    float*    adst  = asrc + (size_t)NB * NN;

    gat_phase1<<<NB * NN / 4, 256, 0, stream>>>(feats, W, asw, adw, hbuf, asrc, adst);
    gat_phase2<<<NB * (NN / 16), 128, 0, stream>>>(madj, mjob, bidx, feats, lam,
                                                   hbuf, asrc, adst, out);
}
